// Qwen2MoEMLPLayer_3530463117600
// MI455X (gfx1250) — matrix operand order repaired and checked by reading, not compiled, not run on hardware
//
#include <hip/hip_runtime.h>
#include <hip/hip_bf16.h>

// ---------------------------------------------------------------------------
// Qwen2-style MoE MLP for MI455X (gfx1250, wave32).
// Compute-bound (~207 GFLOP vs ~0.45 GB traffic) => bf16 WMMA path
// (v_wmma_f32_16x16x32_bf16, 8x FLOP/inst vs fp32 16x16x4 WMMA), f32 accum.
// This revision: loop-invariant gather indices, branch-free masked vector
// loads, float4 + ds_store_b128 fills, register double-buffering so global
// loads overlap WMMA instead of serializing on s_wait_loadcnt 0.
// ---------------------------------------------------------------------------

typedef __bf16 bf16_t;
typedef __attribute__((ext_vector_type(16))) __bf16 v16bf;
typedef __attribute__((ext_vector_type(8)))  __bf16 v8bf;
typedef __attribute__((ext_vector_type(8)))  float  v8f;

#define NUM_EXPERTS 64
#define TOP_K 8
#define NUM_TOKENS 8192
#define HID 1024
#define INTER 512

#define BM 64              // block rows (expert regions aligned to BM)
#define BN 32              // block cols (2 wave-cols of 16)
#define BK 32              // == WMMA K for bf16
#define LDSK (BK + 8)      // 40 halves = 80B row stride (16B aligned, no conflicts)

#define MAXP (NUM_TOKENS * TOP_K + NUM_EXPERTS * BM)  // 69632 padded rows max
#define MAX_MTILES (MAXP / BM)                        // 1088

// ---- workspace layout (bytes) ---------------------------------------------
#define WS_COUNTS   0
#define WS_CURSORS  256
#define WS_AOFF     512
#define WS_TOKW     1024
#define WS_TOKE     (1024 + 262144)
#define WS_ROWSRC   525312
#define WS_ROWW     803840
#define WS_HBUF     1082368          // MAXP*512 bf16 = 71303168 B; total ~72.4 MB

// ---------------------------------------------------------------------------
__global__ __launch_bounds__(256) void moe_init(float* __restrict__ out,
                                                int* __restrict__ counts,
                                                int* __restrict__ cursors,
                                                int* __restrict__ row_src,
                                                float* __restrict__ row_w) {
  const int i = blockIdx.x * blockDim.x + threadIdx.x;
  const int stride = gridDim.x * blockDim.x;
  for (int j = i; j < NUM_TOKENS * HID; j += stride) out[j] = 0.f;
  for (int j = i; j < MAXP; j += stride) { row_src[j] = -1; row_w[j] = 0.f; }
  if (i < NUM_EXPERTS) { counts[i] = 0; cursors[i] = 0; }
}

// ---------------------------------------------------------------------------
__global__ __launch_bounds__(64) void moe_router(const float* __restrict__ hidden,
                                                 const float* __restrict__ rw,
                                                 float* __restrict__ tok_w,
                                                 int* __restrict__ tok_e,
                                                 int* __restrict__ counts) {
  const int t = blockIdx.x;
  const int e = threadIdx.x;
  const float* hrow = hidden + (size_t)t * HID;
  float acc = 0.f;
  for (int k = 0; k < HID; ++k) acc = fmaf(hrow[k], rw[k * NUM_EXPERTS + e], acc);

  __shared__ float probs[NUM_EXPERTS];
  probs[e] = acc;
  __syncthreads();
  if (e == 0) {
    float m = -1e30f;
    for (int j = 0; j < NUM_EXPERTS; ++j) m = fmaxf(m, probs[j]);
    for (int j = 0; j < NUM_EXPERTS; ++j) probs[j] = __expf(probs[j] - m);
    int idx[TOP_K]; float wv[TOP_K]; float wsum = 0.f;
#pragma unroll
    for (int j = 0; j < TOP_K; ++j) {
      int am = 0; float best = -1.f;
      for (int q = 0; q < NUM_EXPERTS; ++q)
        if (probs[q] > best) { best = probs[q]; am = q; }
      idx[j] = am; wv[j] = best; wsum += best; probs[am] = -1.f;
    }
    const float rn = 1.f / wsum;
#pragma unroll
    for (int j = 0; j < TOP_K; ++j) {
      tok_w[t * TOP_K + j] = wv[j] * rn;
      tok_e[t * TOP_K + j] = idx[j];
      atomicAdd(&counts[idx[j]], 1);
    }
  }
}

// ---------------------------------------------------------------------------
__global__ void moe_scan(const int* __restrict__ counts, int* __restrict__ aoff) {
  if (blockIdx.x == 0 && threadIdx.x == 0) {
    int acc = 0;
    for (int e = 0; e < NUM_EXPERTS; ++e) {
      aoff[e] = acc;
      acc += (counts[e] + BM - 1) & ~(BM - 1);
    }
    aoff[NUM_EXPERTS] = acc;
  }
}

// ---------------------------------------------------------------------------
__global__ __launch_bounds__(256) void moe_assign(const float* __restrict__ tok_w,
                                                  const int* __restrict__ tok_e,
                                                  const int* __restrict__ aoff,
                                                  int* __restrict__ cursors,
                                                  int* __restrict__ row_src,
                                                  float* __restrict__ row_w) {
  const int g = blockIdx.x * blockDim.x + threadIdx.x;
  if (g >= NUM_TOKENS * TOP_K) return;
  const int e = tok_e[g];
  const int pos = atomicAdd(&cursors[e], 1);
  const int slot = aoff[e] + pos;
  row_src[slot] = g / TOP_K;
  row_w[slot] = tok_w[g];
}

// ---------------------------------------------------------------------------
// Fragment builders, contiguous k = 16*(lane/16) + s for both A and B (two contiguous 16B LDS loads each).
__device__ __forceinline__ v16bf load_a_frag(const bf16_t* rowp, int b1, int b2) {
  const v8bf lo = *reinterpret_cast<const v8bf*>(rowp + b1);
  const v8bf hi = *reinterpret_cast<const v8bf*>(rowp + b2);
  v16bf a;
#pragma unroll
  for (int i = 0; i < 8; ++i) { a[i] = lo[i]; a[i + 8] = hi[i]; }
  return a;
}
__device__ __forceinline__ v16bf load_b_frag(const bf16_t* colp, int b) {
  const v8bf lo = *reinterpret_cast<const v8bf*>(colp + b);
  const v8bf hi = *reinterpret_cast<const v8bf*>(colp + b + 8);
  v16bf v;
#pragma unroll
  for (int i = 0; i < 8; ++i) { v[i] = lo[i]; v[i + 8] = hi[i]; }
  return v;
}
__device__ __forceinline__ v8bf cvt8(float4 a, float4 b, float msk) {
  v8bf h;
  h[0] = (bf16_t)(a.x * msk); h[1] = (bf16_t)(a.y * msk);
  h[2] = (bf16_t)(a.z * msk); h[3] = (bf16_t)(a.w * msk);
  h[4] = (bf16_t)(b.x * msk); h[5] = (bf16_t)(b.y * msk);
  h[6] = (bf16_t)(b.z * msk); h[7] = (bf16_t)(b.w * msk);
  return h;
}

// ---------------------------------------------------------------------------
// GEMM1: X[rows,1024] @ W_gu[e][1024,1024] with gate/up fused, SiLU epilogue.
__global__ __launch_bounds__(256) void moe_gemm_gateup(
    const float* __restrict__ hidden, const float* __restrict__ w_gu,
    const int* __restrict__ row_src, const int* __restrict__ aoff,
    bf16_t* __restrict__ h_buf) {
  __shared__ bf16_t As[BM][LDSK];
  __shared__ bf16_t Bg[BN][LDSK];   // transposed: [n][k]
  __shared__ bf16_t Bu[BN][LDSK];
  __shared__ int s_e;

  const int tid = threadIdx.x;
  const int r0 = blockIdx.x * BM;
  const int n0 = blockIdx.y * BN;
  if (r0 >= aoff[NUM_EXPERTS]) return;

  if (tid == 0) {
    int e = 0;
    for (int q = 0; q < NUM_EXPERTS; ++q)
      if (aoff[q] <= r0 && r0 < aoff[q + 1]) e = q;
    s_e = e;
  }

  // Loop-invariant fill coordinates (hoisted gather index, branch-free mask).
  const int ar = tid >> 2;                 // 0..63
  const int ac = (tid & 3) * 8;            // 0,8,16,24
  const int src = row_src[r0 + ar];
  const float amsk = (src >= 0) ? 1.f : 0.f;
  const float* aptr = hidden + (size_t)(src >= 0 ? src : 0) * HID + ac;

  const int bk = tid >> 3;                 // 0..31
  const int bn = (tid & 7) * 4;            // 0,4,...,28
  __syncthreads();                         // s_e visible
  const float* gptr = w_gu + (size_t)s_e * HID * (2 * INTER)
                           + (size_t)bk * (2 * INTER) + n0 + bn;

  // Prologue prefetch (tile k0 = 0) into registers.
  float4 fa0 = *(const float4*)(aptr);
  float4 fa1 = *(const float4*)(aptr + 4);
  float4 fg  = *(const float4*)(gptr);
  float4 fu  = *(const float4*)(gptr + INTER);

  const int lane = tid & 31;
  const int wv = tid >> 5, wr = wv >> 1, wc = wv & 1;
  const int mrow = 16 * wr + (lane & 15);
  const int ncol = 16 * wc + (lane & 15);
  const int a_b1 = (lane < 16) ? 0 : 16;
  const int a_b2 = (lane < 16) ? 8 : 24;
  const int b_b  = (lane < 16) ? 0 : 16;

  v8f acc_g = {}; v8f acc_u = {};

  for (int k0 = 0; k0 < HID; k0 += BK) {
    // Commit prefetched tile to LDS (packed cvt + one b128 store for A).
    *reinterpret_cast<v8bf*>(&As[ar][ac]) = cvt8(fa0, fa1, amsk);
    Bg[bn + 0][bk] = (bf16_t)fg.x; Bg[bn + 1][bk] = (bf16_t)fg.y;
    Bg[bn + 2][bk] = (bf16_t)fg.z; Bg[bn + 3][bk] = (bf16_t)fg.w;
    Bu[bn + 0][bk] = (bf16_t)fu.x; Bu[bn + 1][bk] = (bf16_t)fu.y;
    Bu[bn + 2][bk] = (bf16_t)fu.z; Bu[bn + 3][bk] = (bf16_t)fu.w;
    __syncthreads();

    // Issue next tile's global loads; they overlap the WMMA work below.
    if (k0 + BK < HID) {
      const float* ap = aptr + (k0 + BK);
      fa0 = *(const float4*)(ap);
      fa1 = *(const float4*)(ap + 4);
      const float* gp = gptr + (size_t)(k0 + BK) * (2 * INTER);
      fg = *(const float4*)(gp);
      fu = *(const float4*)(gp + INTER);
    }

    const v16bf a  = load_a_frag(&As[mrow][0], a_b1, a_b2);
    const v16bf bg = load_b_frag(&Bg[ncol][0], b_b);
    const v16bf bu = load_b_frag(&Bu[ncol][0], b_b);
    acc_g = __builtin_amdgcn_wmma_f32_16x16x32_bf16(false, a, false, bg, (short)0, acc_g, false, false);
    acc_u = __builtin_amdgcn_wmma_f32_16x16x32_bf16(false, a, false, bu, (short)0, acc_u, false, false);
    __syncthreads();
  }

  const int rbase = r0 + 16 * wr + ((lane < 16) ? 0 : 8);
  const int c = n0 + 16 * wc + (lane & 15);
#pragma unroll
  for (int g = 0; g < 8; ++g) {
    const float gv = acc_g[g], uv = acc_u[g];
    const float h = (gv / (1.f + __expf(-gv))) * uv;   // SiLU(gate) * up
    h_buf[(size_t)(rbase + g) * INTER + c] = (bf16_t)h;
  }
}

// ---------------------------------------------------------------------------
// GEMM2: h[rows,512] @ W_d[e][512,1024]; routing-weight scale + f32 atomic
// scatter-add into out.
__global__ __launch_bounds__(256) void moe_gemm_down(
    const bf16_t* __restrict__ h_buf, const float* __restrict__ w_d,
    const int* __restrict__ row_src, const float* __restrict__ row_w,
    const int* __restrict__ aoff, float* __restrict__ out) {
  __shared__ bf16_t As[BM][LDSK];
  __shared__ bf16_t Bt[BN][LDSK];   // transposed: [n][k]
  __shared__ int s_src[BM];
  __shared__ float s_w[BM];
  __shared__ int s_e;

  const int tid = threadIdx.x;
  const int r0 = blockIdx.x * BM;
  const int n0 = blockIdx.y * BN;
  if (r0 >= aoff[NUM_EXPERTS]) return;

  if (tid == 0) {
    int e = 0;
    for (int q = 0; q < NUM_EXPERTS; ++q)
      if (aoff[q] <= r0 && r0 < aoff[q + 1]) e = q;
    s_e = e;
  }
  if (tid < BM) { s_src[tid] = row_src[r0 + tid]; s_w[tid] = row_w[r0 + tid]; }

  const int ar = tid >> 2;                 // 0..63
  const int ac = (tid & 3) * 8;            // halves
  const bf16_t* aptr = h_buf + (size_t)(r0 + ar) * INTER + ac;

  const int bk = tid >> 3;                 // 0..31
  const int bn = (tid & 7) * 4;
  __syncthreads();                         // s_e / s_src / s_w visible
  const float* bptr = w_d + (size_t)s_e * INTER * HID + (size_t)bk * HID + n0 + bn;

  v8bf fa = *reinterpret_cast<const v8bf*>(aptr);   // A already bf16
  float4 fb = *(const float4*)(bptr);

  const int lane = tid & 31;
  const int wv = tid >> 5, wr = wv >> 1, wc = wv & 1;
  const int mrow = 16 * wr + (lane & 15);
  const int ncol = 16 * wc + (lane & 15);
  const int a_b1 = (lane < 16) ? 0 : 16;
  const int a_b2 = (lane < 16) ? 8 : 24;
  const int b_b  = (lane < 16) ? 0 : 16;

  v8f acc = {};

  for (int k0 = 0; k0 < INTER; k0 += BK) {
    *reinterpret_cast<v8bf*>(&As[ar][ac]) = fa;
    Bt[bn + 0][bk] = (bf16_t)fb.x; Bt[bn + 1][bk] = (bf16_t)fb.y;
    Bt[bn + 2][bk] = (bf16_t)fb.z; Bt[bn + 3][bk] = (bf16_t)fb.w;
    __syncthreads();

    if (k0 + BK < INTER) {
      fa = *reinterpret_cast<const v8bf*>(aptr + (k0 + BK));
      fb = *(const float4*)(bptr + (size_t)(k0 + BK) * HID);
    }

    const v16bf a = load_a_frag(&As[mrow][0], a_b1, a_b2);
    const v16bf b = load_b_frag(&Bt[ncol][0], b_b);
    acc = __builtin_amdgcn_wmma_f32_16x16x32_bf16(false, a, false, b, (short)0, acc, false, false);
    __syncthreads();
  }

  const int lr0 = 16 * wr + ((lane < 16) ? 0 : 8);
  const int c = n0 + 16 * wc + (lane & 15);
#pragma unroll
  for (int g = 0; g < 8; ++g) {
    const int lr = lr0 + g;
    const int tok = s_src[lr];
    if (tok >= 0) atomicAdd(&out[(size_t)tok * HID + c], acc[g] * s_w[lr]);
  }
}

// ---------------------------------------------------------------------------
extern "C" void kernel_launch(void* const* d_in, const int* in_sizes, int n_in,
                              void* d_out, int out_size, void* d_ws, size_t ws_size,
                              hipStream_t stream) {
  const float* hidden = (const float*)d_in[0];   // [8192, 1024]
  const float* rw     = (const float*)d_in[1];   // [1024, 64]
  const float* w_gu   = (const float*)d_in[2];   // [64, 1024, 1024]
  const float* w_d    = (const float*)d_in[3];   // [64, 512, 1024]
  float* out = (float*)d_out;                    // [8192, 1024]

  char* ws = (char*)d_ws;
  int*    counts  = (int*)(ws + WS_COUNTS);
  int*    cursors = (int*)(ws + WS_CURSORS);
  int*    aoff    = (int*)(ws + WS_AOFF);
  float*  tok_w   = (float*)(ws + WS_TOKW);
  int*    tok_e   = (int*)(ws + WS_TOKE);
  int*    row_src = (int*)(ws + WS_ROWSRC);
  float*  row_w   = (float*)(ws + WS_ROWW);
  bf16_t* h_buf   = (bf16_t*)(ws + WS_HBUF);

  moe_init<<<2048, 256, 0, stream>>>(out, counts, cursors, row_src, row_w);
  moe_router<<<NUM_TOKENS, 64, 0, stream>>>(hidden, rw, tok_w, tok_e, counts);
  moe_scan<<<1, 64, 0, stream>>>(counts, aoff);
  moe_assign<<<(NUM_TOKENS * TOP_K) / 256, 256, 0, stream>>>(tok_w, tok_e, aoff,
                                                             cursors, row_src, row_w);
  moe_gemm_gateup<<<dim3(MAX_MTILES, INTER / BN), 256, 0, stream>>>(hidden, w_gu,
                                                                    row_src, aoff, h_buf);
  moe_gemm_down<<<dim3(MAX_MTILES, HID / BN), 256, 0, stream>>>(h_buf, w_d, row_src,
                                                                row_w, aoff, out);
}
